// DeepRelativeST_56049323213814
// MI455X (gfx1250) — compile-verified, matrix operand order checked by reading, not run on hardware
//
#include <hip/hip_runtime.h>

// ---------------------------------------------------------------------------
// Model constants (from the reference)
// ---------------------------------------------------------------------------
constexpr int NB   = 48;          // batch / attended axis
constexpr int CIN  = 32;
constexpr int HW   = 32;
constexpr int OHW  = 30;
constexpr int D    = 256;
constexpr int S    = OHW * OHW;   // 900
constexpr int NH   = 8;
constexpr int DEP  = 32;
constexpr int DFF  = 1024;
constexpr int MACT = NB * S;      // 43200 rows of the (b,900,256)-flattened activations
constexpr float FMAXV = 3.402823466e38f;

// ---------------------------------------------------------------------------
// WMMA types (CDNA5 / gfx1250, wave32)
// ---------------------------------------------------------------------------
typedef __attribute__((ext_vector_type(16))) __bf16 v16bf;
typedef __attribute__((ext_vector_type(8)))  float  v8f;
typedef __attribute__((ext_vector_type(4)))  int    v4i;

struct U128 { unsigned int x, y, z, w; };
union Frag { v16bf v; U128 q[2]; };
union Acc  { v8f v; float f[8]; };

// Optional CDNA5 async copy (GLOBAL_LOAD_ASYNC_TO_LDS_B128, ASYNCcnt)
#if __has_builtin(__builtin_amdgcn_global_load_async_to_lds_b128) && \
    __has_builtin(__builtin_amdgcn_s_wait_asynccnt)
#define HAVE_ASYNC_LDS 1
typedef __attribute__((address_space(1))) v4i* gv4p;   // global int4*
typedef __attribute__((address_space(3))) v4i* lv4p;   // LDS int4*
#else
#define HAVE_ASYNC_LDS 0
#endif

__device__ __forceinline__ unsigned short f32_to_bf16(float f) {
  unsigned int u = __float_as_uint(f);
  u += 0x7FFFu + ((u >> 16) & 1u);           // round-to-nearest-even
  return (unsigned short)(u >> 16);
}

// 16-byte tile-stage helper: global (generic) -> LDS
__device__ __forceinline__ void stage16(unsigned short* lds, const unsigned short* g) {
#if HAVE_ASYNC_LDS
  __builtin_amdgcn_global_load_async_to_lds_b128((gv4p)(g), (lv4p)(lds), 0, 0);
#else
  *(U128*)lds = *(const U128*)g;
#endif
}

__device__ __forceinline__ void stage_wait() {
#if HAVE_ASYNC_LDS
  __builtin_amdgcn_s_wait_asynccnt(0);
#endif
}

// ---------------------------------------------------------------------------
// f32 -> bf16 convert
// ---------------------------------------------------------------------------
__global__ void cvt_bf16_k(const float* __restrict__ in, unsigned short* __restrict__ out, int n) {
  int i = blockIdx.x * blockDim.x + threadIdx.x;
  if (i < n) out[i] = f32_to_bf16(in[i]);
}

// ---------------------------------------------------------------------------
// Generic bf16 WMMA GEMM:  C[M x N] = A[M x K] * B[N x K]^T  (+bias,+relu)
//   GATHER 0: A is bf16 row-major (lda = K)
//   GATHER 1: implicit-GEMM conv1 (valid 3x3, input X bf16 (48,32,32,32), K=288)
//   GATHER 2: implicit-GEMM conv2 (pad-1 3x3, input (43200,256) bf16 rm, K=2304)
//   PERM   : epilogue stores at flat (b, n, s) order = reference's raw reshape
// Block: 128 threads = 4 waves in a 2(M) x 2(N) grid.
// Block tile 64(M) x 128(N) x 32(K); each wave 32x64 -> 8 WMMAs / K-step.
// ---------------------------------------------------------------------------
template <int GATHER, bool BIAS, bool RELU, bool WF32, bool WBF16, bool PERM>
__global__ __launch_bounds__(128) void gemm_wmma(
    const unsigned short* __restrict__ A,
    const unsigned short* __restrict__ Bw,
    const float* __restrict__ bias,
    float* __restrict__ Cf,
    unsigned short* __restrict__ Cb,
    int N, int K)
{
  __shared__ __align__(16) unsigned short lA[64 * 32];    // 4 KB
  __shared__ __align__(16) unsigned short lB[128 * 32];   // 8 KB

  const int tid  = threadIdx.x;
  const int wave = tid >> 5;
  const int lane = tid & 31;
  const int wm   = wave & 1;         // wave M index (0..1)
  const int wn   = wave >> 1;        // wave N index (0..1)
  const int m0 = blockIdx.x * 64;
  const int n0 = blockIdx.y * 128;

  Acc acc[2][4];
#pragma unroll
  for (int p = 0; p < 2; ++p)
#pragma unroll
    for (int t = 0; t < 4; ++t)
#pragma unroll
      for (int e = 0; e < 8; ++e) acc[p][t].f[e] = 0.f;

  // ---- staging assignments ----
  const int rA  = tid >> 1;          // A: row 0..63, 32B (2x16B) per thread
  const int hhA = (tid & 1) * 16;    // A: K half (0 or 16)
  // B: row tid (0..127), 64B (4x16B) per thread

  // direct-path global cursors (advance by 32 ushorts per K-step)
  const unsigned short* srcA = A + (size_t)(m0 + rA) * K + hhA;
  const unsigned short* srcB = Bw + (size_t)(n0 + tid) * K;

  // gather-path per-thread invariants
  int g_bb = 0, g_y = 0, g_x = 0;
  size_t g_base = 0;
  if constexpr (GATHER != 0) {
    const int m = m0 + rA;
    g_bb = m / S;
    const int ss = m - g_bb * S;
    g_y = ss / OHW;
    g_x = ss - g_y * OHW;
    if constexpr (GATHER == 1) g_base = (size_t)g_bb * CIN * HW * HW;
  }

  for (int k0 = 0; k0 < K; k0 += 32) {
    // ---- stage A tile ----
    if constexpr (GATHER == 0) {
      stage16(&lA[rA * 32 + hhA],     srcA);
      stage16(&lA[rA * 32 + hhA + 8], srcA + 8);
      if (k0 + 32 < K) __builtin_prefetch(srcA + 32, 0, 1);   // global_prefetch_b8
      srcA += 32;
    } else {
      // incremental k -> (ci, ky, kx): one division per 16-element chunk
      int k  = k0 + hhA;
      int ci = k / 9;
      int rem = k - ci * 9;
      int ky = rem / 3;
      int kx = rem - ky * 3;
#pragma unroll
      for (int e = 0; e < 16; ++e) {
        unsigned short v;
        if constexpr (GATHER == 1) {
          // conv1: valid 3x3; (y+ky)<32, (x+kx)<32 always in bounds
          v = A[g_base + (size_t)ci * (HW * HW) + (g_y + ky) * HW + (g_x + kx)];
        } else {
          const int yy = g_y + ky - 1, xx = g_x + kx - 1;
          v = (yy >= 0 && yy < OHW && xx >= 0 && xx < OHW)
                ? A[((size_t)(g_bb * S + yy * OHW + xx)) * D + ci]
                : (unsigned short)0;
        }
        lA[rA * 32 + hhA + e] = v;
        // advance (kx,ky,ci) branchlessly
        ++kx;
        const int c3 = (kx == 3); kx = c3 ? 0 : kx; ky += c3;
        const int c9 = (ky == 3); ky = c9 ? 0 : ky; ci += c9;
      }
    }
    // ---- stage B tile (weights, always N x K row-major) ----
    {
#pragma unroll
      for (int qq = 0; qq < 4; ++qq)
        stage16(&lB[tid * 32 + qq * 8], srcB + qq * 8);
      if (k0 + 32 < K) __builtin_prefetch(srcB + 32, 0, 1);
      srcB += 32;
    }
    stage_wait();
    __syncthreads();

    // ---- fragments (ISA 16-bit layout: lanes<16 K0-7/16-23, lanes>=16 K8-15/24-31)
    const int lrow = lane & 15;
    const int ka0  = (lane < 16) ? 0 : 8;
    const int ka1  = (lane < 16) ? 16 : 24;
    Frag a[2];
#pragma unroll
    for (int p = 0; p < 2; ++p) {
      const int ar = wm * 32 + p * 16 + lrow;
      a[p].q[0] = *(const U128*)&lA[ar * 32 + ka0];
      a[p].q[1] = *(const U128*)&lA[ar * 32 + ka1];
    }
#pragma unroll
    for (int t = 0; t < 4; ++t) {
      Frag b;
      const int br = wn * 64 + t * 16 + lrow;
      b.q[0] = *(const U128*)&lB[br * 32 + ka0];
      b.q[1] = *(const U128*)&lB[br * 32 + ka1];
#pragma unroll
      for (int p = 0; p < 2; ++p)
        acc[p][t].v = __builtin_amdgcn_wmma_f32_16x16x32_bf16(
            false, a[p].v, false, b.v, (short)0, acc[p][t].v, false, false);
    }
    __syncthreads();
  }

  // ---- epilogue (C layout: lanes<16 -> M=r, lanes>=16 -> M=8+r; N = lane&15) ----
  const int lrow = lane & 15;
  const int mo   = (lane < 16) ? 0 : 8;
#pragma unroll
  for (int p = 0; p < 2; ++p) {
#pragma unroll
    for (int t = 0; t < 4; ++t) {
#pragma unroll
      for (int rr = 0; rr < 8; ++rr) {
        const int m = m0 + wm * 32 + p * 16 + mo + rr;
        const int n = n0 + wn * 64 + t * 16 + lrow;
        float v = acc[p][t].f[rr];
        if constexpr (BIAS) v += bias[n];
        if constexpr (RELU) v = fmaxf(v, 0.f);
        size_t idx;
        if constexpr (PERM) {
          const int bb = m / S, ss = m - bb * S;
          idx = (size_t)bb * (D * S) + (size_t)n * S + ss;   // (b, c, y, x) flat
        } else {
          idx = (size_t)m * N + n;
        }
        if constexpr (WF32)  Cf[idx] = v;
        if constexpr (WBF16) Cb[idx] = f32_to_bf16(v);
      }
    }
  }
}

// ---------------------------------------------------------------------------
// scores[s,h,i,j] = (q_i . k_j)/sqrt(32) ; emd[s,h,i,j] = q_i . relw[s,h,:,j]
// Q/K flat (43200,256) rm; row = s*48+i, cols h*32..h*32+31. One block per (s,h).
// ---------------------------------------------------------------------------
__global__ __launch_bounds__(128) void scores_emd_k(
    const float* __restrict__ Q, const float* __restrict__ Kp,
    const float* __restrict__ relw,
    float* __restrict__ scores, float* __restrict__ emd)
{
  const int s = blockIdx.x, h = blockIdx.y;
  const int tid = threadIdx.x;
  __shared__ float qt[48 * 32], kt[48 * 32], wt[32 * 48];
  for (int idx = tid; idx < 48 * 32; idx += 128) {
    const int i = idx >> 5, d = idx & 31;
    const size_t g = (size_t)(s * 48 + i) * D + h * 32 + d;
    qt[idx] = Q[g];
    kt[idx] = Kp[g];
  }
  for (int idx = tid; idx < 32 * 48; idx += 128) {
    const int d = idx / 48, j = idx % 48;
    wt[idx] = relw[(((size_t)s * NH + h) * DEP + d) * 48 + j];
  }
  __syncthreads();
  const float rs = 0.17677669529663687f;   // 1/sqrt(32)
  const size_t base = ((size_t)s * NH + h) * 2304;
  for (int o = tid; o < 2304; o += 128) {
    const int i = o / 48, j = o % 48;
    float a = 0.f, e = 0.f;
#pragma unroll
    for (int d = 0; d < 32; ++d) {
      const float qv = qt[i * 32 + d];
      a += qv * kt[j * 32 + d];
      e += qv * wt[d * 48 + j];
    }
    scores[base + o] = a * rs;
    emd[base + o]    = e;
  }
}

// ---------------------------------------------------------------------------
// Decoder masked path: qk1 = scores@l1w.T + b1; bmm[j,i] = (qk1^T @ l2w.T + b2) + mask
// ---------------------------------------------------------------------------
__global__ __launch_bounds__(128) void masked_tx_k(
    const float* __restrict__ scores,
    const float* __restrict__ l1w, const float* __restrict__ l1b,
    const float* __restrict__ l2w, const float* __restrict__ l2b,
    float* __restrict__ bmm)
{
  const int s = blockIdx.x, h = blockIdx.y;
  const int tid = threadIdx.x;
  __shared__ float sc[2304], q1[2304], w1[2304], w2[2304];
  const size_t base = ((size_t)s * NH + h) * 2304;
  for (int o = tid; o < 2304; o += 128) {
    sc[o] = scores[base + o];
    w1[o] = l1w[o];
    w2[o] = l2w[o];
  }
  __syncthreads();
  for (int o = tid; o < 2304; o += 128) {          // qk1[i,j]
    const int i = o / 48, j = o % 48;
    float a = l1b[j];
#pragma unroll 8
    for (int jp = 0; jp < 48; ++jp) a += sc[i * 48 + jp] * w1[j * 48 + jp];
    q1[o] = a;
  }
  __syncthreads();
  for (int o = tid; o < 2304; o += 128) {          // bmm[j,i]
    const int j = o / 48, i = o % 48;
    float a = l2b[i];
#pragma unroll 8
    for (int ip = 0; ip < 48; ++ip) a += q1[ip * 48 + j] * w2[i * 48 + ip];
    if (i > j) a -= FMAXV;                          // causal mask (triu k=1)
    bmm[base + o] = a;
  }
}

// ---------------------------------------------------------------------------
// Softmax over the S axis (dim 0) of (S,NH,48,48), with skew gather folded in.
// In place: bmm -> attn.  skew[s,h,a,b] = emd[s,h,r,c] (t=a*95+47+b) if c<48.
// ---------------------------------------------------------------------------
__global__ void softmax_s_k(float* __restrict__ bmm, const float* __restrict__ emd)
{
  const int col = blockIdx.x * blockDim.x + threadIdx.x;   // (h, a, b)
  if (col >= NH * 2304) return;
  const int h = col / 2304;
  const int a = (col / 48) % 48, b = col % 48;
  const int t = a * 95 + 47 + b;
  const int rr = t / 96, cc = t % 96;
  const bool sk = (cc < 48);
  const size_t eoff   = (size_t)h * 2304 + rr * 48 + cc;
  const size_t stride = (size_t)NH * 2304;                 // 18432

  float mx = -INFINITY;
  for (int s = 0; s < S; ++s) {
    float v = bmm[(size_t)s * stride + col];
    if (sk) v += emd[(size_t)s * stride + eoff];
    mx = fmaxf(mx, v);
  }
  float sum = 0.f;
  for (int s = 0; s < S; ++s) {
    float v = bmm[(size_t)s * stride + col];
    if (sk) v += emd[(size_t)s * stride + eoff];
    sum += __expf(v - mx);
  }
  const float inv = 1.f / sum;
  for (int s = 0; s < S; ++s) {
    float v = bmm[(size_t)s * stride + col];
    if (sk) v += emd[(size_t)s * stride + eoff];
    bmm[(size_t)s * stride + col] = __expf(v - mx) * inv;
  }
}

// ---------------------------------------------------------------------------
// out[s,h,i,d] = sum_j attn[s,h,i,j] * V[s*48+j, h*32+d]   (flat (s,h,i,d))
// ---------------------------------------------------------------------------
__global__ __launch_bounds__(128) void attn_v_k(
    const float* __restrict__ attn, const float* __restrict__ V,
    float* __restrict__ out)
{
  const int s = blockIdx.x, h = blockIdx.y;
  const int tid = threadIdx.x;
  __shared__ float at[2304], vt[48 * 32];
  const size_t abase = ((size_t)s * NH + h) * 2304;
  for (int o = tid; o < 2304; o += 128) at[o] = attn[abase + o];
  for (int o = tid; o < 48 * 32; o += 128) {
    const int j = o >> 5, d = o & 31;
    vt[o] = V[(size_t)(s * 48 + j) * D + h * 32 + d];
  }
  __syncthreads();
  const size_t obase = ((size_t)s * NH + h) * 1536;
  for (int o = tid; o < 1536; o += 128) {
    const int i = o >> 5, d = o & 31;
    float a = 0.f;
#pragma unroll 8
    for (int j = 0; j < 48; ++j) a += at[i * 48 + j] * vt[j * 32 + d];
    out[obase + o] = a;
  }
}

// ---------------------------------------------------------------------------
// Residual + LayerNorm over rows of 256. One wave (32 lanes) per row.
// ---------------------------------------------------------------------------
__global__ __launch_bounds__(256) void add_ln_k(
    const float* __restrict__ xa, const float* __restrict__ xb,
    float* __restrict__ outF, unsigned short* __restrict__ outB)
{
  const int wave = threadIdx.x >> 5, lane = threadIdx.x & 31;
  const size_t row  = (size_t)blockIdx.x * 8 + wave;
  const size_t base = row * D;
  float v[8];
  float sum = 0.f;
#pragma unroll
  for (int e = 0; e < 8; ++e) {
    const size_t i = base + e * 32 + lane;
    v[e] = xa[i] + xb[i];
    sum += v[e];
  }
  for (int off = 16; off; off >>= 1) sum += __shfl_xor(sum, off, 32);
  const float mean = sum * (1.f / 256.f);
  float var = 0.f;
#pragma unroll
  for (int e = 0; e < 8; ++e) { const float d = v[e] - mean; var += d * d; }
  for (int off = 16; off; off >>= 1) var += __shfl_xor(var, off, 32);
  const float inv = rsqrtf(var * (1.f / 256.f) + 1e-5f);
#pragma unroll
  for (int e = 0; e < 8; ++e) {
    const size_t i = base + e * 32 + lane;
    const float o = (v[e] - mean) * inv;
    if (outF) outF[i] = o;
    if (outB) outB[i] = f32_to_bf16(o);
  }
}

// ---------------------------------------------------------------------------
// y[row] = h3[row,:] . out_w + out_b   (one wave per row)
// ---------------------------------------------------------------------------
__global__ __launch_bounds__(256) void final_proj_k(
    const float* __restrict__ h3, const float* __restrict__ w,
    const float* __restrict__ b, float* __restrict__ y)
{
  const int wave = threadIdx.x >> 5, lane = threadIdx.x & 31;
  const size_t row = (size_t)blockIdx.x * 8 + wave;
  float s = 0.f;
#pragma unroll
  for (int e = 0; e < 8; ++e) {
    const int c = e * 32 + lane;
    s += h3[row * D + c] * w[c];
  }
  for (int off = 16; off; off >>= 1) s += __shfl_xor(s, off, 32);
  if (lane == 0) y[row] = s + b[0];
}

// softmax over the batch (48) axis of y (48,900) -> d_out
__global__ void final_softmax_k(const float* __restrict__ y, float* __restrict__ out)
{
  const int ss = blockIdx.x * blockDim.x + threadIdx.x;
  if (ss >= S) return;
  float mx = -INFINITY;
  for (int bb = 0; bb < NB; ++bb) mx = fmaxf(mx, y[(size_t)bb * S + ss]);
  float sum = 0.f;
  for (int bb = 0; bb < NB; ++bb) sum += __expf(y[(size_t)bb * S + ss] - mx);
  const float inv = 1.f / sum;
  for (int bb = 0; bb < NB; ++bb)
    out[(size_t)bb * S + ss] = __expf(y[(size_t)bb * S + ss] - mx) * inv;
}

// ---------------------------------------------------------------------------
// Host driver
// ---------------------------------------------------------------------------
extern "C" void kernel_launch(void* const* d_in, const int* in_sizes, int n_in,
                              void* d_out, int out_size, void* d_ws, size_t ws_size,
                              hipStream_t stream)
{
  (void)in_sizes; (void)n_in; (void)out_size; (void)ws_size;

  const float* X_en    = (const float*)d_in[0];
  const float* X_de    = (const float*)d_in[1];
  const float* conv1_w = (const float*)d_in[2];
  const float* conv1_b = (const float*)d_in[3];
  const float* conv2_w = (const float*)d_in[4];
  const float* conv2_b = (const float*)d_in[5];
  const float* enc_wq  = (const float*)d_in[6];
  const float* enc_wk  = (const float*)d_in[7];
  const float* enc_wv  = (const float*)d_in[8];
  const float* enc_rel = (const float*)d_in[9];
  const float* enc_f1  = (const float*)d_in[10];
  const float* enc_fb1 = (const float*)d_in[11];
  const float* enc_f2  = (const float*)d_in[12];
  const float* enc_fb2 = (const float*)d_in[13];
  const float* dm_wq   = (const float*)d_in[14];
  const float* dm_wk   = (const float*)d_in[15];
  const float* dm_wv   = (const float*)d_in[16];
  const float* dm_rel  = (const float*)d_in[17];
  const float* l1w     = (const float*)d_in[18];
  const float* l1b     = (const float*)d_in[19];
  const float* l2w     = (const float*)d_in[20];
  const float* l2b     = (const float*)d_in[21];
  const float* dc_wq   = (const float*)d_in[22];
  const float* dc_wk   = (const float*)d_in[23];
  const float* dc_wv   = (const float*)d_in[24];
  const float* dc_rel  = (const float*)d_in[25];
  const float* dc_f1   = (const float*)d_in[26];
  const float* dc_fb1  = (const float*)d_in[27];
  const float* dc_f2   = (const float*)d_in[28];
  const float* dc_fb2  = (const float*)d_in[29];
  const float* out_w   = (const float*)d_in[30];
  const float* out_b   = (const float*)d_in[31];
  float* OUT = (float*)d_out;

  // ---- workspace bump allocator (deterministic layout every call) ----
  char* base = (char*)d_ws;
  size_t cur = 0;
  auto bump = [&](size_t bytes) -> void* {
    void* p = base + cur;
    cur += (bytes + 255) & ~(size_t)255;
    return p;
  };
  auto fbuf = [&](size_t n) { return (float*)bump(n * sizeof(float)); };
  auto hbuf = [&](size_t n) { return (unsigned short*)bump(n * sizeof(unsigned short)); };

  // bf16 weights
  unsigned short* Wc1  = hbuf(256 * 288);
  unsigned short* Wc2  = hbuf(256 * 2304);
  unsigned short* Weq  = hbuf(D * D);
  unsigned short* Wek  = hbuf(D * D);
  unsigned short* Wev  = hbuf(D * D);
  unsigned short* Wmq  = hbuf(D * D);
  unsigned short* Wmk  = hbuf(D * D);
  unsigned short* Wmv  = hbuf(D * D);
  unsigned short* Wcq  = hbuf(D * D);
  unsigned short* Wck  = hbuf(D * D);
  unsigned short* Wcv  = hbuf(D * D);
  unsigned short* Wef1 = hbuf(DFF * D);
  unsigned short* Wef2 = hbuf(D * DFF);
  unsigned short* Wdf1 = hbuf(DFF * D);
  unsigned short* Wdf2 = hbuf(D * DFF);
  // bf16 inputs / intermediates
  unsigned short* Xen_bf = hbuf((size_t)NB * CIN * HW * HW);
  unsigned short* Xde_bf = hbuf((size_t)NB * CIN * HW * HW);
  unsigned short* C1_bf  = hbuf((size_t)MACT * D);         // conv1 out, row-major
  unsigned short* B_xen  = hbuf((size_t)MACT * D);
  unsigned short* B_xde  = hbuf((size_t)MACT * D);
  unsigned short* B_enc  = hbuf((size_t)MACT * D);
  unsigned short* B_h1   = hbuf((size_t)MACT * D);
  unsigned short* B_h2   = hbuf((size_t)MACT * D);
  unsigned short* FFmid  = hbuf((size_t)MACT * DFF);
  // f32 activations
  float* F_xen = fbuf((size_t)MACT * D);
  float* F_xde = fbuf((size_t)MACT * D);
  float* F_enc = fbuf((size_t)MACT * D);
  float* F_q   = fbuf((size_t)MACT * D);
  float* F_k   = fbuf((size_t)MACT * D);
  float* F_v   = fbuf((size_t)MACT * D);
  float* F_att = fbuf((size_t)MACT * D);
  float* F_h1  = fbuf((size_t)MACT * D);
  float* F_h2  = fbuf((size_t)MACT * D);
  float* F_tmp = fbuf((size_t)MACT * D);
  // attention tensors (S,NH,48,48)
  float* Sc = fbuf((size_t)S * NH * 2304);
  float* Se = fbuf((size_t)S * NH * 2304);
  float* Sb = fbuf((size_t)S * NH * 2304);
  float* Yv = fbuf((size_t)MACT);

  auto cvt = [&](const float* src, unsigned short* dst, size_t n) {
    cvt_bf16_k<<<dim3((unsigned)((n + 255) / 256)), 256, 0, stream>>>(src, dst, (int)n);
  };

  // ---- phase 0: convert weights + inputs to bf16 ----
  cvt(conv1_w, Wc1, 256 * 288);
  cvt(conv2_w, Wc2, 256 * 2304);
  cvt(enc_wq, Weq, D * D);  cvt(enc_wk, Wek, D * D);  cvt(enc_wv, Wev, D * D);
  cvt(dm_wq, Wmq, D * D);   cvt(dm_wk, Wmk, D * D);   cvt(dm_wv, Wmv, D * D);
  cvt(dc_wq, Wcq, D * D);   cvt(dc_wk, Wck, D * D);   cvt(dc_wv, Wcv, D * D);
  cvt(enc_f1, Wef1, (size_t)DFF * D);  cvt(enc_f2, Wef2, (size_t)D * DFF);
  cvt(dc_f1, Wdf1, (size_t)DFF * D);   cvt(dc_f2, Wdf2, (size_t)D * DFF);
  cvt(X_en, Xen_bf, (size_t)NB * CIN * HW * HW);
  cvt(X_de, Xde_bf, (size_t)NB * CIN * HW * HW);

  const dim3 gD(MACT / 64, D / 128);     // 675 x 2 (N = 256)
  const dim3 gF(MACT / 64, DFF / 128);   // 675 x 8 (N = 1024)
  const dim3 gSH(S, NH);

  auto conv_pair = [&](const unsigned short* Xbf, float* Fx, unsigned short* Bx) {
    gemm_wmma<1, true, false, false, true, false><<<gD, 128, 0, stream>>>(
        Xbf, Wc1, conv1_b, nullptr, C1_bf, D, CIN * 9);
    gemm_wmma<2, true, false, true, true, true><<<gD, 128, 0, stream>>>(
        C1_bf, Wc2, conv2_b, Fx, Bx, D, D * 9);
  };
  auto proj = [&](const unsigned short* Abf, const unsigned short* W, float* Cf) {
    gemm_wmma<0, false, false, true, false, false><<<gD, 128, 0, stream>>>(
        Abf, W, nullptr, Cf, nullptr, D, D);
  };
  auto ffn = [&](const unsigned short* Abf, const unsigned short* W1, const float* b1,
                 const unsigned short* W2, const float* b2, float* Cf) {
    gemm_wmma<0, true, true, false, true, false><<<gF, 128, 0, stream>>>(
        Abf, W1, b1, nullptr, FFmid, DFF, D);
    gemm_wmma<0, true, false, true, false, false><<<gD, 128, 0, stream>>>(
        FFmid, W2, b2, Cf, nullptr, D, DFF);
  };
  auto addln = [&](const float* xa, const float* xb, float* of, unsigned short* ob) {
    add_ln_k<<<dim3(MACT / 8), 256, 0, stream>>>(xa, xb, of, ob);
  };

  // ---- phase 1: convs (encoder & decoder streams) ----
  conv_pair(Xen_bf, F_xen, B_xen);
  conv_pair(Xde_bf, F_xde, B_xde);

  // ---- phase 2: encoder self-attention + FFN ----
  proj(B_xen, Weq, F_q);  proj(B_xen, Wek, F_k);  proj(B_xen, Wev, F_v);
  scores_emd_k<<<gSH, 128, 0, stream>>>(F_q, F_k, enc_rel, Sc, Se);
  softmax_s_k<<<dim3(NH * 2304 / 256), 256, 0, stream>>>(Sc, Se);
  attn_v_k<<<gSH, 128, 0, stream>>>(Sc, F_v, F_att);
  addln(F_xen, F_att, F_h1, B_h1);                       // o1
  ffn(B_h1, Wef1, enc_fb1, Wef2, enc_fb2, F_tmp);
  addln(F_h1, F_tmp, F_enc, B_enc);                      // enc_out

  // ---- phase 3: decoder masked self-attention ----
  proj(B_xde, Wmq, F_q);  proj(B_xde, Wmk, F_k);  proj(B_xde, Wmv, F_v);
  scores_emd_k<<<gSH, 128, 0, stream>>>(F_q, F_k, dm_rel, Sc, Se);
  masked_tx_k<<<gSH, 128, 0, stream>>>(Sc, l1w, l1b, l2w, l2b, Sb);
  softmax_s_k<<<dim3(NH * 2304 / 256), 256, 0, stream>>>(Sb, Se);
  attn_v_k<<<gSH, 128, 0, stream>>>(Sb, F_v, F_att);
  addln(F_xde, F_att, F_h1, B_h1);                       // h1

  // ---- phase 4: decoder cross-attention ----
  proj(B_h1, Wcq, F_q);  proj(B_enc, Wck, F_k);  proj(B_enc, Wcv, F_v);
  scores_emd_k<<<gSH, 128, 0, stream>>>(F_q, F_k, dc_rel, Sc, Se);
  softmax_s_k<<<dim3(NH * 2304 / 256), 256, 0, stream>>>(Sc, Se);
  attn_v_k<<<gSH, 128, 0, stream>>>(Sc, F_v, F_att);
  addln(F_h1, F_att, F_h2, B_h2);                        // h2 = LN(c + h1)

  // ---- phase 5: decoder FFN ----
  ffn(B_h2, Wdf1, dc_fb1, Wdf2, dc_fb2, F_tmp);
  addln(F_h2, F_tmp, F_q, nullptr);                      // h3 (reuse F_q)

  // ---- phase 6: output projection + softmax over batch axis ----
  final_proj_k<<<dim3(MACT / 8), 256, 0, stream>>>(F_q, out_w, out_b, Yv);
  final_softmax_k<<<dim3((S + 255) / 256), 256, 0, stream>>>(Yv, OUT);
}
